// MeshConvolution_30769145708867
// MI455X (gfx1250) — matrix operand order repaired and checked by reading, not compiled, not run on hardware
//
#include <hip/hip_runtime.h>

typedef __attribute__((ext_vector_type(16))) _Float16 v16h;
typedef __attribute__((ext_vector_type(8)))  _Float16 v8h;
typedef __attribute__((ext_vector_type(4)))  _Float16 v4h;
typedef __attribute__((ext_vector_type(8)))  float    v8f;
typedef __attribute__((ext_vector_type(4)))  int      v4i;

#define CCH 32           // channels in/out
#define KTAPS 5
#define KTOT 160         // 5*32 contraction length
#define IMG_STRIDE 168   // halfs; 336B rows: 16B aligned, spreads banks
#define KT_STRIDE 168
#define WPB 8            // waves per 256-thread block

__device__ __forceinline__ float4 f4add(float4 a, float4 b) {
    return make_float4(a.x + b.x, a.y + b.y, a.z + b.z, a.w + b.w);
}
__device__ __forceinline__ float4 f4absub(float4 a, float4 b) {
    return make_float4(fabsf(a.x - b.x), fabsf(a.y - b.y),
                       fabsf(a.z - b.z), fabsf(a.w - b.w));
}
__device__ __forceinline__ v4h f4cvt(float4 a) {
    v4h r;
    r[0] = (_Float16)a.x; r[1] = (_Float16)a.y;
    r[2] = (_Float16)a.z; r[3] = (_Float16)a.w;
    return r;
}

__global__ __launch_bounds__(256) void meshconv_wmma(
    const float* __restrict__ features,      // [E,32]
    const int*   __restrict__ nbrs,          // [E,4]
    const float* __restrict__ kern,          // [5,32,32] -> flat [160][32]
    const float* __restrict__ bias,          // [32]
    float*       __restrict__ out,           // [E,32]
    int E, int nTiles)
{
    __shared__ _Float16 kT[CCH * KT_STRIDE];              // kT[n][k] (transposed, f16)
    __shared__ _Float16 img[WPB * 16 * IMG_STRIDE];       // per-wave 16x160 image tiles

    const int tid  = threadIdx.x;
    const int lane = tid & 31;
    const int wave = tid >> 5;

    // ---- stage kernel matrix into LDS: f32 -> f16, transposed ----
    for (int i = tid; i < KTOT * CCH; i += 256) {
        int kk = i >> 5;       // 0..159 (tap*32 + c_in)
        int n  = i & 31;       // c_out
        kT[n * KT_STRIDE + kk] = (_Float16)kern[i];
    }
    __syncthreads();

    const int ncol = lane & 15;     // output column within 16-wide tile
    const int hi16 = lane >> 4;     // 0 = lanes 0-15, 1 = lanes 16-31

    // ---- B fragments: resident in VGPRs for whole kernel ----
    // ISA B layout (16b, K=32): lane group 0 -> K base 0, group 1 -> K base 8;
    // elems 0..7 at base, elems 8..15 at base+16. Column N = lane&15.
    v16h Bfrag[KTAPS][2];
    #pragma unroll
    for (int kb = 0; kb < KTAPS; ++kb) {
        #pragma unroll
        for (int nt = 0; nt < 2; ++nt) {
            const _Float16* p = &kT[(nt * 16 + ncol) * KT_STRIDE + kb * 32 + hi16 * 8];
            v8h lo = *(const v8h*)(p);
            v8h hi = *(const v8h*)(p + 16);
            Bfrag[kb][nt] = __builtin_shufflevector(lo, hi,
                0,1,2,3,4,5,6,7,8,9,10,11,12,13,14,15);
        }
    }
    const float b0 = bias[ncol];
    const float b1 = bias[16 + ncol];

    _Float16* myimg = &img[wave * 16 * IMG_STRIDE];
    const int erow = lane >> 1;          // edge row (0..15) this lane builds
    const int cg   = (lane & 1) * 16;    // channel group (0 or 16)
    const int gstride = gridDim.x * WPB;

    for (int t = blockIdx.x * WPB + wave; t < nTiles; t += gstride) {
        const int base = t * 16;

        // ---------- build A (image rows) into this wave's LDS slice ----------
        {
            int e = base + erow;
            if (e >= E) e = E - 1;
            // index array is streamed once -> non-temporal load (don't pollute L2)
            const v4i idx = __builtin_nontemporal_load((const v4i*)(nbrs + 4 * (long)e));
            const float4* f0p = (const float4*)(features + (long)e      * CCH + cg);
            const float4* n0p = (const float4*)(features + (long)idx[0] * CCH + cg);
            const float4* n1p = (const float4*)(features + (long)idx[1] * CCH + cg);
            const float4* n2p = (const float4*)(features + (long)idx[2] * CCH + cg);
            const float4* n3p = (const float4*)(features + (long)idx[3] * CCH + cg);
            _Float16* rowp = myimg + erow * IMG_STRIDE + cg;
            #pragma unroll
            for (int j = 0; j < 4; ++j) {
                float4 f0 = f0p[j];
                float4 a0 = n0p[j], a1 = n1p[j], a2 = n2p[j], a3 = n3p[j];
                *(v4h*)(rowp + 0 * 32 + 4 * j) = f4cvt(f0);
                *(v4h*)(rowp + 1 * 32 + 4 * j) = f4cvt(f4add(a0, a2));
                *(v4h*)(rowp + 2 * 32 + 4 * j) = f4cvt(f4add(a1, a3));
                *(v4h*)(rowp + 3 * 32 + 4 * j) = f4cvt(f4absub(a0, a2));
                *(v4h*)(rowp + 4 * 32 + 4 * j) = f4cvt(f4absub(a1, a3));
            }
        }

        // per-wave sync: all lanes' ds_stores visible before ds_loads below
        asm volatile("s_wait_dscnt 0x0" ::: "memory");
        __builtin_amdgcn_wave_barrier();

        // ---------- WMMA: acc[16x16] x2, K = 5 steps of 32 ----------
        v8f acc0, acc1;
        #pragma unroll
        for (int r = 0; r < 8; ++r) { acc0[r] = b0; acc1[r] = b1; }

        // ISA A layout (16b, 16x32): lane group 0 -> K base 0, group 1 -> K base 8;
        // elems 0..7 at base, elems 8..15 at base+16. Row M = lane&15.
        const _Float16* arow = myimg + (lane & 15) * IMG_STRIDE + hi16 * 8;
        #pragma unroll
        for (int kb = 0; kb < KTAPS; ++kb) {
            const _Float16* p = arow + kb * 32;
            v8h lo = *(const v8h*)(p);
            v8h hi = *(const v8h*)(p + 16);
            v16h a = __builtin_shufflevector(lo, hi,
                0,1,2,3,4,5,6,7,8,9,10,11,12,13,14,15);
            acc0 = __builtin_amdgcn_wmma_f32_16x16x32_f16(
                false, a, false, Bfrag[kb][0], (short)0, acc0, false, false);
            acc1 = __builtin_amdgcn_wmma_f32_16x16x32_f16(
                false, a, false, Bfrag[kb][1], (short)0, acc1, false, false);
        }

        // ---------- store: row M = r + 8*hi16, cols ncol and ncol+16 ----------
        // Output is write-once/never-read: non-temporal stores keep the 128MB
        // feature table resident in the 192MB L2 for the gathers.
        #pragma unroll
        for (int r = 0; r < 8; ++r) {
            int e = base + r + hi16 * 8;
            if (e < E) {
                float* po = out + (long)e * CCH + ncol;
                __builtin_nontemporal_store(acc0[r], po);
                __builtin_nontemporal_store(acc1[r], po + 16);
            }
        }
    }
}

extern "C" void kernel_launch(void* const* d_in, const int* in_sizes, int n_in,
                              void* d_out, int out_size, void* d_ws, size_t ws_size,
                              hipStream_t stream) {
    const float* features = (const float*)d_in[0];
    const int*   nbrs     = (const int*)d_in[1];
    const float* kern     = (const float*)d_in[2];
    const float* bias     = (const float*)d_in[3];
    float* out = (float*)d_out;

    const int E = in_sizes[0] / CCH;
    const int nTiles = (E + 15) / 16;
    int blocks = (nTiles + WPB - 1) / WPB;
    if (blocks > 2048) blocks = 2048;
    if (blocks < 1) blocks = 1;

    meshconv_wmma<<<blocks, 256, 0, stream>>>(features, nbrs, kern, bias, out, E, nTiles);
}
